// Network_5574867550480
// MI455X (gfx1250) — matrix operand order repaired and checked by reading, not compiled, not run on hardware
//
#include <hip/hip_runtime.h>
#include <stdint.h>

// ---------------------------------------------------------------------------
// Types for WMMA fragments (gfx1250, wave32)
// ---------------------------------------------------------------------------
typedef __attribute__((ext_vector_type(16))) _Float16 v16h;
typedef __attribute__((ext_vector_type(8)))  float    v8f;
typedef __attribute__((ext_vector_type(4)))  uint32_t u32x4;
typedef __attribute__((ext_vector_type(4)))  float    f32x4;

// Load a 16-f16 fragment as two 16-byte chunks (elements 0..7 from p0, 8..15 from p1).
__device__ __forceinline__ v16h load2x8h(const _Float16* p0, const _Float16* p1) {
    union { v16h v; u32x4 u[2]; } r;
    r.u[0] = *(const u32x4*)p0;
    r.u[1] = *(const u32x4*)p1;
    return r.v;
}

// ---------------------------------------------------------------------------
// One dense MLP layer: dst[n][m] = relu( sum_k W[m][k] * src[n][k] + bias[m] )
// src/dst are LDS, row-major [128 points][256 channels] f16 (stride 256).
// W is global f16, row-major [256][KC*32].
// 8 waves: wave w covers output channels [32w, 32w+32); each wave does
// 2 M-tiles x 8 N-tiles of 16x16 WMMA output.
//
// Fragment layouts per CDNA5 ISA (CDNA5 ISA):
//  A 16x32 f16 : lane<16  -> row M=lane,    elems 0..7 = K 0..7,  8..15 = K 16..23
//                lane>=16 -> row M=lane-16, elems 0..7 = K 8..15, 8..15 = K 24..31
//  B 32x16 f16 : lane<16  -> col N=lane,    elems 0..7 = K 0..7,  8..15 = K 16..23
//                lane>=16 -> col N=lane-16, elems 0..7 = K 8..15, 8..15 = K 24..31
//  D 16x16 f32 : lane<16  -> col N=lane,    vgpr r = M r
//                lane>=16 -> col N=lane-16, vgpr r = M r+8
// ---------------------------------------------------------------------------
template<int KC, bool RELU>
__device__ __forceinline__ void mlp_layer(const _Float16* __restrict__ src,
                                          _Float16*       __restrict__ dst,
                                          const _Float16* __restrict__ wgt,
                                          const float*    __restrict__ bias) {
    const int tid  = threadIdx.x;
    const int wave = tid >> 5;
    const int lane = tid & 31;
    const int r    = lane & 15;
    const int hi   = lane >> 4;
    const int Kpad = KC * 32;

    for (int mt = 0; mt < 2; ++mt) {
        const int mbase = (wave * 2 + mt) * 16;

        // A fragments for this M-tile, all K chunks (weights: L1/L2 resident)
        v16h a[KC];
        const _Float16* wrow = wgt + (size_t)(mbase + r) * Kpad + hi * 8;
#pragma unroll
        for (int kc = 0; kc < KC; ++kc)
            a[kc] = load2x8h(wrow + kc * 32, wrow + kc * 32 + 16);

        // Bias for this lane's 8 output channels: c = mbase + hi*8 + i
        float bv[8];
        const float* bp = bias + mbase + hi * 8;
#pragma unroll
        for (int i = 0; i < 8; ++i) bv[i] = bp[i];

#pragma unroll
        for (int nt = 0; nt < 8; ++nt) {
            const int nbase = nt * 16;
            v8f acc = {};
            const _Float16* srow = src + (nbase + r) * 256 + hi * 8;
#pragma unroll
            for (int kc = 0; kc < KC; ++kc) {
                v16h b = load2x8h(srow + kc * 32, srow + kc * 32 + 16);
                acc = __builtin_amdgcn_wmma_f32_16x16x32_f16(
                          false, a[kc], false, b, (short)0, acc, false, false);
            }
            // bias + relu + f16 pack, one b128 store per lane
            union { _Float16 h[8]; u32x4 u; } o;
#pragma unroll
            for (int i = 0; i < 8; ++i) {
                float v = acc[i] + bv[i];
                if (RELU) v = v > 0.0f ? v : 0.0f;
                o.h[i] = (_Float16)v;
            }
            *(u32x4*)(dst + (nbase + r) * 256 + mbase + hi * 8) = o.u;
        }
    }
}

// ---------------------------------------------------------------------------
// Bilinear sample of 24 channels of one plane at (u,v) in [0,1],
// align_corners=True on a 512x512 grid. Writes f16 into LDS.
//   chbase : first triplane channel (plane*48 + half*24)
//   dst    : LDS destination for these 24 channels
// ---------------------------------------------------------------------------
__device__ __forceinline__ void sample_plane24(const float* __restrict__ tp,
                                               float u, float v, int chbase,
                                               _Float16* __restrict__ dst) {
    const int RES = 512;
    float ix = u * 511.0f, iy = v * 511.0f;
    float fx = floorf(ix), fy = floorf(iy);
    float wx1 = ix - fx, wy1 = iy - fy;
    float wx0 = 1.0f - wx1, wy0 = 1.0f - wy1;
    int x0 = min(max((int)fx, 0), 511);
    int x1 = min(max((int)fx + 1, 0), 511);
    int y0 = min(max((int)fy, 0), 511);
    int y1 = min(max((int)fy + 1, 0), 511);
    float w00 = wy0 * wx0, w01 = wy0 * wx1, w10 = wy1 * wx0, w11 = wy1 * wx1;
    int i00 = y0 * RES + x0, i01 = y0 * RES + x1;
    int i10 = y1 * RES + x0, i11 = y1 * RES + x1;
    const float* base = tp + (size_t)chbase * RES * RES;
#pragma unroll 4
    for (int c = 0; c < 24; ++c) {
        const float* pc = base + (size_t)c * RES * RES;
        float val = pc[i00] * w00 + pc[i01] * w01 + pc[i10] * w10 + pc[i11] * w11;
        dst[c] = (_Float16)val;
    }
}

// ---------------------------------------------------------------------------
// Weight pack: f32 -> f16, K padded to multiple of 32 for layer 0.
// ---------------------------------------------------------------------------
__global__ void pack_weights_kernel(const float* __restrict__ w0,
                                    const float* __restrict__ w1,
                                    const float* __restrict__ w2,
                                    _Float16* __restrict__ wp0,
                                    _Float16* __restrict__ wp1,
                                    _Float16* __restrict__ wp2) {
    int i = blockIdx.x * blockDim.x + threadIdx.x;
    if (i < 256 * 160) {                       // wp0: 256 x 160 (144 real + 16 zero)
        int o = i / 160, c = i % 160;
        wp0[i] = (c < 144) ? (_Float16)w0[o * 144 + c] : (_Float16)0.0f;
    }
    if (i < 256 * 256) {                       // wp1/wp2: 256 x 256
        wp1[i] = (_Float16)w1[i];
        wp2[i] = (_Float16)w2[i];
    }
}

// ---------------------------------------------------------------------------
// Fused kernel: triplane sampling -> LDS -> 3 WMMA layers -> alpha
// 256 threads (8 wave32), 128 points per block, 128 KB dynamic LDS.
// ---------------------------------------------------------------------------
__global__ __launch_bounds__(256)
void triplane_mlp_kernel(const float* __restrict__ points,
                         const float* __restrict__ triplanes,
                         const _Float16* __restrict__ wp0, const float* __restrict__ b0,
                         const _Float16* __restrict__ wp1, const float* __restrict__ b1,
                         const _Float16* __restrict__ wp2, const float* __restrict__ b2,
                         const float* __restrict__ wa, const float* __restrict__ ba,
                         float* __restrict__ out) {
    extern __shared__ char smem[];
    _Float16* actA = (_Float16*)smem;                 // [128][256] f16 = 64 KB
    _Float16* actB = actA + 128 * 256;                // [128][256] f16 = 64 KB

    const int tid   = threadIdx.x;
    const int pbase = blockIdx.x * 128;

    // ---- Phase 0: stage this block's 128x3 point coords into LDS (coalesced)
    // pts overlays actB (dead until layer 0 writes it, after a barrier).
    float* pts = (float*)actB;                        // 1536 B
    {
        const f32x4* src4 = (const f32x4*)(points + (size_t)pbase * 3);
        f32x4*       dst4 = (f32x4*)pts;
        if (tid < 96) dst4[tid] = src4[tid];          // 384 floats = 96 x float4
    }
    __syncthreads();

    // ---- Phase 1: sample 3 planes x 48 channels per point into actA ----
    // feats row layout: [0,48) = xy(plane0 @ x,y), [48,96) = yz(plane1 @ y,z),
    // [96,144) = xz(plane2 @ x,z), [144,160) = zero pad, [160,256) unused.
    // 768 tasks (point x plane x half) = exactly 3 full passes of 256 threads.
    for (int task = tid; task < 128 * 6; task += 256) {
        const int p    = task / 6;
        const int rem  = task % 6;
        const int pl   = rem >> 1;                    // plane 0..2
        const int half = rem & 1;                     // 24-channel half
        const float px = pts[p * 3 + 0];
        const float py = pts[p * 3 + 1];
        const float pz = pts[p * 3 + 2];
        float u, v;
        if      (pl == 0) { u = px; v = py; }
        else if (pl == 1) { u = py; v = pz; }
        else              { u = px; v = pz; }
        const int c0 = pl * 48 + half * 24;
        sample_plane24(triplanes, u, v, c0, actA + p * 256 + c0);
        if (rem == 0) {                               // zero pad [144,160)
#pragma unroll
            for (int c = 144; c < 160; ++c) actA[p * 256 + c] = (_Float16)0.0f;
        }
    }
    __syncthreads();

    // ---- Phase 2: MLP (WMMA f32 <- f16 x f16) ----
    mlp_layer<5, true>(actA, actB, wp0, b0);   // 144(pad160) -> 256
    __syncthreads();
    mlp_layer<8, true>(actB, actA, wp1, b1);   // 256 -> 256
    __syncthreads();
    mlp_layer<8, true>(actA, actB, wp2, b2);   // 256 -> 256
    __syncthreads();

    // ---- Phase 3: alpha = wa . act + ba (all 256 threads, vector loads) ----
    // psum overlays actA (dead: layer 2 read it before the barrier above).
    float* psum = (float*)actA;
    {
        const int p    = tid & 127;
        const int half = tid >> 7;                    // channels [half*128, half*128+128)
        const _Float16* arow = actB + p * 256 + half * 128;
        const float*    wah  = wa + half * 128;
        float s = 0.0f;
#pragma unroll
        for (int i = 0; i < 16; ++i) {                // 16 x 8 f16 = 128 channels
            union { u32x4 u; _Float16 h[8]; } av;
            av.u = *(const u32x4*)(arow + i * 8);
            f32x4 w0v = *(const f32x4*)(wah + i * 8);
            f32x4 w1v = *(const f32x4*)(wah + i * 8 + 4);
            s += (float)av.h[0] * w0v.x + (float)av.h[1] * w0v.y +
                 (float)av.h[2] * w0v.z + (float)av.h[3] * w0v.w +
                 (float)av.h[4] * w1v.x + (float)av.h[5] * w1v.y +
                 (float)av.h[6] * w1v.z + (float)av.h[7] * w1v.w;
        }
        psum[tid] = s;
    }
    __syncthreads();
    if (tid < 128) {
        float alpha = psum[tid] + psum[tid + 128] + ba[0];
        __builtin_nontemporal_store(alpha, out + pbase + tid);
    }
}

// ---------------------------------------------------------------------------
// Launch
// ---------------------------------------------------------------------------
extern "C" void kernel_launch(void* const* d_in, const int* in_sizes, int n_in,
                              void* d_out, int out_size, void* d_ws, size_t ws_size,
                              hipStream_t stream) {
    const float* points    = (const float*)d_in[0];   // (1, 524288, 3)
    const float* triplanes = (const float*)d_in[1];   // (144, 512, 512)
    const float* w0        = (const float*)d_in[2];   // (256, 144)
    const float* b0        = (const float*)d_in[3];   // (256,)
    const float* w1        = (const float*)d_in[4];   // (256, 256)
    const float* b1        = (const float*)d_in[5];   // (256,)
    const float* w2        = (const float*)d_in[6];   // (256, 256)
    const float* b2        = (const float*)d_in[7];   // (256,)
    const float* wa        = (const float*)d_in[8];   // (1, 256)
    const float* ba        = (const float*)d_in[9];   // (1,)
    float* out             = (float*)d_out;           // 524288 floats

    // Workspace: packed f16 weights
    char* ws = (char*)d_ws;
    _Float16* wp0 = (_Float16*)ws;                         // 256*160*2 = 81920 B
    _Float16* wp1 = (_Float16*)(ws + 81920);               // 256*256*2 = 131072 B
    _Float16* wp2 = (_Float16*)(ws + 81920 + 131072);      // 131072 B

    pack_weights_kernel<<<(256 * 256 + 255) / 256, 256, 0, stream>>>(
        w0, w1, w2, wp0, wp1, wp2);

    const int NPTS   = 524288;
    const int blocks = NPTS / 128;                 // 4096
    const size_t lds = 2u * 128u * 256u * sizeof(_Float16);  // 128 KB
    triplane_mlp_kernel<<<blocks, 256, lds, stream>>>(
        points, triplanes, wp0, b0, wp1, b1, wp2, b2, wa, ba, out);
}
